// GNNModel_54477365183080
// MI455X (gfx1250) — matrix operand order repaired and checked by reading, not compiled, not run on hardware
//
#include <hip/hip_runtime.h>
#include <hip/hip_bf16.h>

// ---------------------------------------------------------------------------
// Types for CDNA5 WMMA (wave32)
// ---------------------------------------------------------------------------
typedef __attribute__((ext_vector_type(16))) __bf16 bf16x16;
typedef __attribute__((ext_vector_type(4)))  __bf16 bf16x4;
typedef __attribute__((ext_vector_type(2)))  __bf16 bf16x2;
typedef __attribute__((ext_vector_type(8)))  float  floatx8;

#define NNODES 16384
#define PG     256      // nodes per graph
#define NB     64       // graphs
#define HDIM   512
#define CDIM   12
#define INDIM  4096
#define ODIM   5

// ---------------------------------------------------------------------------
// Tiled WMMA GEMM:  C = act( A[M,K] @ B[K,N] + bias ) (optionally += C)
// A, B f32 in global; converted to bf16 into double-buffered LDS.
// Block tile 128x128, K-step 32, 8 waves; each wave owns 32x64 = 2x4 wmma
// tiles -> 8 v_wmma_f32_16x16x32_bf16 per K-step.
// Requires: N % 128 == 0, K % 32 == 0, lda % 4 == 0. M arbitrary (bounded).
// ---------------------------------------------------------------------------
#define TM 128
#define TN 128
#define TK 32
#define LDA_S 36            // 32 + 4 pad (bf16 units)
#define LDB_S 36

// A: 128x32 tile = 1024 float4 slots (4/thread), vectorized both sides.
// B: 128(N)x32(K) tile, transposed into LDS; each thread owns 8 K-pairs so
//    the LDS store is a packed bf16x2 (b32); global stays coalesced along N.
#define LOAD_TILE_REGS(k0)                                                   \
    {                                                                        \
        _Pragma("unroll")                                                    \
        for (int u = 0; u < 4; u++) {                                        \
            int i = tid + u * 256;                                           \
            int r = i >> 3, cq = (i & 7) * 4;                                \
            int gr = blockM + r;                                             \
            if (gr < M)                                                      \
                ra4[u] = *(const float4*)&A[(long long)gr * lda + (k0) + cq];\
            else                                                             \
                ra4[u] = make_float4(0.f, 0.f, 0.f, 0.f);                    \
        }                                                                    \
        _Pragma("unroll")                                                    \
        for (int u = 0; u < 8; u++) {                                        \
            int i = tid + u * 256;                                           \
            int kp = i >> 7, n = i & 127;                                    \
            const float* bp = &B[(long long)(2 * kp + (k0)) * ldb + (blockN + n)]; \
            rb0[u] = bp[0];                                                  \
            rb1[u] = bp[ldb];                                                \
        }                                                                    \
    }

template <bool BIAS, bool ACC, bool RELU>
__global__ void __launch_bounds__(256)
gemm_bf16_wmma(const float* __restrict__ A, const float* __restrict__ B,
               float* __restrict__ C,
               int M, int N, int K, int lda, int ldb, int ldc,
               long long strideA, long long strideB, long long strideC,
               const float* __restrict__ bias)
{
    __shared__ __bf16 sA[2 * TM * LDA_S];
    __shared__ __bf16 sB[2 * TN * LDB_S];

    const int bz = blockIdx.z;
    A += (long long)bz * strideA;
    B += (long long)bz * strideB;
    C += (long long)bz * strideC;

    const int tid   = threadIdx.x;
    const int wid   = tid >> 5;
    const int lane  = tid & 31;
    const int waveM = wid >> 1;    // 0..3 : 32 rows each
    const int waveN = wid & 1;     // 0..1 : 64 cols each
    const int hlf   = lane >> 4;   // 0 or 1
    const int lrow  = lane & 15;

    const int blockM = blockIdx.y * TM;
    const int blockN = blockIdx.x * TN;

    floatx8 acc[2][4] = {};
    float4 ra4[4];
    float  rb0[8], rb1[8];

    const int nk = K / TK;

    // preload tile 0 into registers
    LOAD_TILE_REGS(0);

    for (int kt = 0; kt < nk; kt++) {
        const int cur = kt & 1;
        __bf16* sAc = sA + cur * (TM * LDA_S);
        __bf16* sBc = sB + cur * (TN * LDB_S);

        // stage registers -> LDS (f32 -> bf16, packed vector stores)
        #pragma unroll
        for (int u = 0; u < 4; u++) {
            int i = tid + u * 256;
            int r = i >> 3, cq = (i & 7) * 4;
            bf16x4 p;
            p[0] = (__bf16)ra4[u].x; p[1] = (__bf16)ra4[u].y;
            p[2] = (__bf16)ra4[u].z; p[3] = (__bf16)ra4[u].w;
            *(bf16x4*)&sAc[r * LDA_S + cq] = p;
        }
        #pragma unroll
        for (int u = 0; u < 8; u++) {
            int i = tid + u * 256;
            int kp = i >> 7, n = i & 127;
            bf16x2 p;
            p[0] = (__bf16)rb0[u]; p[1] = (__bf16)rb1[u];
            *(bf16x2*)&sBc[n * LDB_S + 2 * kp] = p;
        }
        __syncthreads();

        // issue global loads for next tile early (overlap with wmma below)
        if (kt + 1 < nk) { LOAD_TILE_REGS((kt + 1) * TK); }

        // --- fragments per ISA 7.12.2 layouts ---
        bf16x16 afrag[2], bfrag[4];
        #pragma unroll
        for (int i = 0; i < 2; i++) {
            const __bf16* base =
                &sAc[(waveM * 32 + i * 16 + lrow) * LDA_S + hlf * 8];
            #pragma unroll
            for (int j = 0; j < 8; j++) {
                afrag[i][j]     = base[j];        // K = hlf*8 + j
                afrag[i][8 + j] = base[16 + j];   // K = 16 + hlf*8 + j
            }
        }
        #pragma unroll
        for (int j = 0; j < 4; j++) {
            const __bf16* base =
                &sBc[(waveN * 64 + j * 16 + lrow) * LDB_S + hlf * 8];
            #pragma unroll
            for (int q = 0; q < 16; q++)
                bfrag[j][q] = base[q < 8 ? q : 8 + q];  // K = hlf*8 + q (q<8), 16 + hlf*8 + (q-8)
        }

        #pragma unroll
        for (int i = 0; i < 2; i++)
            #pragma unroll
            for (int j = 0; j < 4; j++)
                acc[i][j] = __builtin_amdgcn_wmma_f32_16x16x32_bf16(
                    false, afrag[i], false, bfrag[j],
                    (short)0, acc[i][j], false, false);
        __syncthreads();
    }

    // --- epilogue: D element v -> row v + 8*hlf, col lane&15 ---
    #pragma unroll
    for (int i = 0; i < 2; i++) {
        #pragma unroll
        for (int j = 0; j < 4; j++) {
            const int col = blockN + waveN * 64 + j * 16 + lrow;
            float bval = 0.0f;
            if (BIAS) bval = bias[col];
            #pragma unroll
            for (int v = 0; v < 8; v++) {
                int row = blockM + waveM * 32 + i * 16 + v + (hlf ? 8 : 0);
                if (row < M) {
                    float val = acc[i][j][v] + bval;
                    long long idx = (long long)row * ldc + col;
                    if (ACC)  val += C[idx];
                    if (RELU) val = fmaxf(val, 0.0f);
                    C[idx] = val;
                }
            }
        }
    }
}

// ---------------------------------------------------------------------------
// Small helper kernels
// ---------------------------------------------------------------------------
__global__ void zero_f32(float* p, int n) {
    int i = blockIdx.x * 256 + threadIdx.x;
    if (i < n) p[i] = 0.0f;
}

// A_raw[b][s][d] += 1 for each directed edge (list already symmetrized)
__global__ void build_adj(const int* __restrict__ ei, float* __restrict__ A, int E) {
    int e = blockIdx.x * 256 + threadIdx.x;
    if (e >= E) return;
    int s = ei[e];
    int d = ei[E + e];
    int b = s >> 8;
    atomicAdd(&A[((long long)b << 16) + ((s & 255) << 8) + (d & 255)], 1.0f);
}

// deg[i] = 1 + column-sum of A_b ; dinv = deg^-1/2
__global__ void deg_dinv(const float* __restrict__ A, float* __restrict__ dinv) {
    int i = blockIdx.x * 256 + threadIdx.x;
    if (i >= NNODES) return;
    int b = i >> 8, d = i & 255;
    const float* Ab = A + ((long long)b << 16);
    float s = 1.0f;                 // self-loop
    for (int r = 0; r < PG; r++) s += Ab[r * PG + d];
    dinv[i] = 1.0f / sqrtf(s);
}

// An[b][d][s] = dinv[d]*dinv[s]*A[b][s][d] + (s==d)*dinv[d]^2
__global__ void build_An(const float* __restrict__ A, const float* __restrict__ dinv,
                         float* __restrict__ An) {
    int i = blockIdx.x * 256 + threadIdx.x;      // up to 64*256*256 = 16.7M
    if (i >= NB * PG * PG) return;
    int b = i >> 16;
    int d = (i >> 8) & 255;
    int s = i & 255;
    float dd = dinv[(b << 8) + d];
    float v  = A[((long long)b << 16) + (s << 8) + d] * dinv[(b << 8) + s] * dd;
    if (s == d) v += dd * dd;
    An[i] = v;
}

// sel = softmax(T1 @ Wm2 + bm2) over C=12 ; one wave per node
__global__ void sel_softmax(const float* __restrict__ T1, const float* __restrict__ Wm2,
                            const float* __restrict__ bm2, float* __restrict__ sel) {
    int gw   = (blockIdx.x * blockDim.x + threadIdx.x) >> 5;
    int lane = threadIdx.x & 31;
    if (gw >= NNODES) return;
    const float* t = T1 + (long long)gw * HDIM;
    float part[CDIM];
    #pragma unroll
    for (int c = 0; c < CDIM; c++) part[c] = 0.0f;
    for (int h = lane; h < HDIM; h += 32) {
        float tv = t[h];
        const float* w = Wm2 + h * CDIM;
        #pragma unroll
        for (int c = 0; c < CDIM; c++) part[c] += tv * w[c];
    }
    #pragma unroll
    for (int c = 0; c < CDIM; c++) {
        float v = part[c];
        for (int off = 16; off; off >>= 1) v += __shfl_down(v, off, 32);
        part[c] = v;
    }
    if (lane == 0) {
        float mx = -1e30f;
        #pragma unroll
        for (int c = 0; c < CDIM; c++) { part[c] += bm2[c]; mx = fmaxf(mx, part[c]); }
        float s = 0.0f, e[CDIM];
        #pragma unroll
        for (int c = 0; c < CDIM; c++) { e[c] = expf(part[c] - mx); s += e[c]; }
        float inv = 1.0f / s;
        #pragma unroll
        for (int c = 0; c < CDIM; c++) sel[(long long)gw * CDIM + c] = e[c] * inv;
    }
}

// AS[b][n][k] = sum_m A[b][n][m] * sel[b][m][k]
__global__ void compute_AS(const float* __restrict__ A, const float* __restrict__ sel,
                           float* __restrict__ AS) {
    int i = blockIdx.x * 256 + threadIdx.x;
    if (i >= NNODES * CDIM) return;
    int c = i % CDIM, n = i / CDIM;
    int b = n >> 8, nl = n & 255;
    const float* Ar = A + ((long long)b << 16) + (nl << 8);
    const float* sb = sel + (long long)b * PG * CDIM;
    float s = 0.0f;
    for (int m = 0; m < PG; m++) s += Ar[m] * sb[m * CDIM + c];
    AS[i] = s;
}

__global__ void row_degrees(const float* __restrict__ A, float* __restrict__ degs) {
    int n = blockIdx.x * 256 + threadIdx.x;
    if (n >= NNODES) return;
    int b = n >> 8, nl = n & 255;
    const float* Ar = A + ((long long)b << 16) + (nl << 8);
    float s = 0.0f;
    for (int m = 0; m < PG; m++) s += Ar[m];
    degs[n] = s;
}

// out[b][c][h] = selu( sum_n sel[b][n][c] * H[b][n][h] )
__global__ void pool_out(const float* __restrict__ sel, const float* __restrict__ H,
                         float* __restrict__ outp) {
    int i = blockIdx.x * 256 + threadIdx.x;
    if (i >= NB * CDIM * HDIM) return;
    int h = i & 511;
    int c = (i >> 9) % CDIM;
    int b = i / (CDIM * HDIM);
    const float* sb = sel + (long long)b * PG * CDIM + c;
    const float* Hb = H + (long long)b * PG * HDIM + h;
    float s = 0.0f;
    for (int n = 0; n < PG; n++) s += sb[n * CDIM] * Hb[n * HDIM];
    const float lam = 1.0507009873554805f, alp = 1.6732632423543772f;
    outp[i] = (s > 0.0f) ? lam * s : lam * alp * (expf(s) - 1.0f);
}

// out_adj[b][c][k] = sum_n sel[b][n][c] * AS[b][n][k]
__global__ void compute_outadj(const float* __restrict__ sel, const float* __restrict__ AS,
                               float* __restrict__ oadj) {
    int i = blockIdx.x * 256 + threadIdx.x;
    if (i >= NB * CDIM * CDIM) return;
    int k = i % CDIM, c = (i / CDIM) % CDIM, b = i / (CDIM * CDIM);
    const float* sb = sel + (long long)b * PG * CDIM;
    const float* ab = AS + (long long)b * PG * CDIM;
    float s = 0.0f;
    for (int n = 0; n < PG; n++) s += sb[n * CDIM + c] * ab[n * CDIM + k];
    oadj[i] = s;
}

// per-graph: DMoN losses + zero-diag + sym degree-normalize of out_adj
__global__ void __launch_bounds__(256)
loss_and_norm(const float* __restrict__ sel, const float* __restrict__ degs,
              const float* __restrict__ oadj, float* __restrict__ oanorm,
              float* __restrict__ lp) {
    __shared__ float s_ss[CDIM * CDIM];
    __shared__ float s_ca[CDIM];
    __shared__ float s_cs[CDIM];
    __shared__ float s_d[CDIM];
    __shared__ float s_red[256];
    int b = blockIdx.x, t = threadIdx.x;
    const float* sb  = sel  + (long long)b * PG * CDIM;
    const float* db  = degs + b * PG;
    const float* oab = oadj + b * CDIM * CDIM;

    s_red[t] = db[t];
    __syncthreads();
    for (int off = 128; off; off >>= 1) {
        if (t < off) s_red[t] += s_red[t + off];
        __syncthreads();
    }
    float m = s_red[0] * 0.5f;

    if (t < CDIM * CDIM) {
        int c = t / CDIM, k = t % CDIM;
        float s = 0.0f;
        for (int n = 0; n < PG; n++) s += sb[n * CDIM + c] * sb[n * CDIM + k];
        s_ss[t] = s;
    }
    if (t < CDIM) {
        float ca = 0.0f, cs = 0.0f;
        for (int n = 0; n < PG; n++) {
            float sv = sb[n * CDIM + t];
            ca += sv * db[n];
            cs += sv;
        }
        s_ca[t] = ca; s_cs[t] = cs;
        float rs = 0.0f;
        for (int k = 0; k < CDIM; k++) if (k != t) rs += oab[t * CDIM + k];
        s_d[t] = sqrtf(rs) + 1e-15f;
    }
    __syncthreads();
    if (t < CDIM * CDIM) {
        int c = t / CDIM, k = t % CDIM;
        oanorm[b * CDIM * CDIM + t] = (c == k) ? 0.0f : oab[t] / (s_d[k] * s_d[c]);
    }
    if (t == 0) {
        float tr = 0.0f, trn = 0.0f;
        for (int c = 0; c < CDIM; c++) {
            tr  += oab[c * CDIM + c];
            trn += s_ca[c] * s_ca[c];
        }
        trn /= (2.0f * m);
        float spectral = -(tr - trn) / (2.0f * m);
        float ssn = 0.0f;
        for (int i = 0; i < CDIM * CDIM; i++) ssn += s_ss[i] * s_ss[i];
        ssn = sqrtf(ssn);
        const float isqC = 0.28867513459481287f;   // 1/sqrt(12)
        float ortho = 0.0f;
        for (int i = 0; i < CDIM * CDIM; i++) {
            float d = s_ss[i] / ssn - ((i / CDIM == i % CDIM) ? isqC : 0.0f);
            ortho += d * d;
        }
        ortho = sqrtf(ortho);
        float cn = 0.0f;
        for (int c = 0; c < CDIM; c++) cn += s_cs[c] * s_cs[c];
        float cluster = sqrtf(cn) / (float)PG * 3.4641016151377544f - 1.0f;
        lp[b] = spectral + ortho + cluster;
    }
}

// t2[b][c][h] = sum_k oanorm[b][c][k] * out[b][k][h]
__global__ void coarse_conv(const float* __restrict__ oanorm, const float* __restrict__ outp,
                            float* __restrict__ t2) {
    int i = blockIdx.x * 256 + threadIdx.x;
    if (i >= NB * CDIM * HDIM) return;
    int h = i & 511;
    int c = (i >> 9) % CDIM;
    int b = i / (CDIM * HDIM);
    const float* oar = oanorm + b * CDIM * CDIM + c * CDIM;
    const float* ob  = outp + (long long)b * CDIM * HDIM + h;
    float s = 0.0f;
    #pragma unroll
    for (int k = 0; k < CDIM; k++) s += oar[k] * ob[k * HDIM];
    t2[i] = s;
}

__global__ void graph_mean(const float* __restrict__ h2, float* __restrict__ g) {
    int i = blockIdx.x * 256 + threadIdx.x;
    if (i >= NB * HDIM) return;
    int b = i >> 9, h = i & 511;
    const float* hb = h2 + (long long)b * CDIM * HDIM + h;
    float s = 0.0f;
    #pragma unroll
    for (int c = 0; c < CDIM; c++) s += hb[c * HDIM];
    g[i] = s * (1.0f / 12.0f);
}

__global__ void final_logits(const float* __restrict__ g2, const float* __restrict__ Wl2,
                             const float* __restrict__ bl2, float* __restrict__ out) {
    int i = blockIdx.x * 256 + threadIdx.x;
    if (i >= NB * ODIM) return;
    int o = i % ODIM, b = i / ODIM;
    const float* gr = g2 + b * HDIM;
    float s = bl2[o];
    for (int h = 0; h < HDIM; h++) s += gr[h] * Wl2[h * ODIM + o];
    out[i] = s;
}

__global__ void loss_reduce(const float* __restrict__ lp, float* __restrict__ out) {
    __shared__ float r[64];
    int t = threadIdx.x;
    r[t] = lp[t];
    __syncthreads();
    for (int off = 32; off; off >>= 1) {
        if (t < off) r[t] += r[t + off];
        __syncthreads();
    }
    if (t == 0) out[NB * ODIM] = r[0] / (float)NB;
}

// ---------------------------------------------------------------------------
extern "C" void kernel_launch(void* const* d_in, const int* in_sizes, int n_in,
                              void* d_out, int out_size, void* d_ws, size_t ws_size,
                              hipStream_t stream) {
    const float* x     = (const float*)d_in[0];
    const int*   ei    = (const int*)  d_in[1];
    // d_in[2] = batch (unused: graphs are dense, node i -> graph i/256)
    const float* W1    = (const float*)d_in[3];
    const float* b1    = (const float*)d_in[4];
    const float* Wm1   = (const float*)d_in[5];
    const float* bm1   = (const float*)d_in[6];
    const float* Wm2   = (const float*)d_in[7];
    const float* bm2   = (const float*)d_in[8];
    const float* Wrel  = (const float*)d_in[9];
    const float* brel  = (const float*)d_in[10];
    const float* Wroot = (const float*)d_in[11];
    const float* Wl1   = (const float*)d_in[12];
    const float* bl1   = (const float*)d_in[13];
    const float* Wl2   = (const float*)d_in[14];
    const float* bl2   = (const float*)d_in[15];
    float* out = (float*)d_out;

    const int E = in_sizes[1] / 2;                 // 262144 directed edges

    // workspace layout (floats)
    float* ws     = (float*)d_ws;
    float* A_raw  = ws;                            // 64*256*256
    float* An     = A_raw + NB * PG * PG;          // 64*256*256
    float* h0     = An + NB * PG * PG;             // 16384*512 (reused as T1)
    float* H      = h0 + NNODES * HDIM;            // 16384*512
    float* dinv   = H + NNODES * HDIM;             // 16384
    float* sel    = dinv + NNODES;                 // 16384*12
    float* AS     = sel + NNODES * CDIM;           // 16384*12
    float* outp   = AS + NNODES * CDIM;            // 64*12*512
    float* oadj   = outp + NB * CDIM * HDIM;       // 64*144
    float* oanorm = oadj + NB * CDIM * CDIM;       // 64*144
    float* degs   = oanorm + NB * CDIM * CDIM;     // 16384
    float* t2     = degs + NNODES;                 // 64*12*512
    float* h2     = t2 + NB * CDIM * HDIM;         // 64*12*512
    float* g      = h2 + NB * CDIM * HDIM;         // 64*512
    float* g2     = g + NB * HDIM;                 // 64*512
    float* lp     = g2 + NB * HDIM;                // 64

    dim3 blk(256);

    // --- adjacency / normalization ---
    zero_f32<<<(NB * PG * PG + 255) / 256, blk, 0, stream>>>(A_raw, NB * PG * PG);
    build_adj<<<(E + 255) / 256, blk, 0, stream>>>(ei, A_raw, E);
    deg_dinv<<<NNODES / 256, blk, 0, stream>>>(A_raw, dinv);
    build_An<<<(NB * PG * PG + 255) / 256, blk, 0, stream>>>(A_raw, dinv, An);

    // --- h0 = x @ W1 (the big one: 16384x4096x512) ---
    gemm_bf16_wmma<false, false, false>
        <<<dim3(HDIM / TN, NNODES / TM, 1), blk, 0, stream>>>(
        x, W1, h0, NNODES, HDIM, INDIM, INDIM, HDIM, HDIM, 0, 0, 0, nullptr);

    // --- H = relu(An @ h0 + b1), batched over 64 graphs ---
    gemm_bf16_wmma<true, false, true>
        <<<dim3(HDIM / TN, PG / TM, NB), blk, 0, stream>>>(
        An, h0, H, PG, HDIM, PG, PG, HDIM, HDIM,
        (long long)PG * PG, (long long)PG * HDIM, (long long)PG * HDIM, b1);

    // --- T1 = H @ Wm1 + bm1 (reuse h0 buffer) ---
    gemm_bf16_wmma<true, false, false>
        <<<dim3(HDIM / TN, NNODES / TM, 1), blk, 0, stream>>>(
        H, Wm1, h0, NNODES, HDIM, HDIM, HDIM, HDIM, HDIM, 0, 0, 0, bm1);

    // --- sel = softmax(T1 @ Wm2 + bm2) ---
    sel_softmax<<<(NNODES * 32) / 256, blk, 0, stream>>>(h0, Wm2, bm2, sel);

    compute_AS<<<(NNODES * CDIM + 255) / 256, blk, 0, stream>>>(A_raw, sel, AS);
    row_degrees<<<NNODES / 256, blk, 0, stream>>>(A_raw, degs);
    pool_out<<<(NB * CDIM * HDIM + 255) / 256, blk, 0, stream>>>(sel, H, outp);
    compute_outadj<<<(NB * CDIM * CDIM + 255) / 256, blk, 0, stream>>>(sel, AS, oadj);
    loss_and_norm<<<NB, blk, 0, stream>>>(sel, degs, oadj, oanorm, lp);

    // --- DenseGraphConv: h2 = (oanorm@out)@Wrel + brel + out@Wroot ---
    coarse_conv<<<(NB * CDIM * HDIM + 255) / 256, blk, 0, stream>>>(oanorm, outp, t2);
    gemm_bf16_wmma<true, false, false>
        <<<dim3(HDIM / TN, (NB * CDIM + TM - 1) / TM, 1), blk, 0, stream>>>(
        t2, Wrel, h2, NB * CDIM, HDIM, HDIM, HDIM, HDIM, HDIM, 0, 0, 0, brel);
    gemm_bf16_wmma<false, true, false>
        <<<dim3(HDIM / TN, (NB * CDIM + TM - 1) / TM, 1), blk, 0, stream>>>(
        outp, Wroot, h2, NB * CDIM, HDIM, HDIM, HDIM, HDIM, HDIM, 0, 0, 0, nullptr);

    // --- readout ---
    graph_mean<<<(NB * HDIM + 255) / 256, blk, 0, stream>>>(h2, g);
    gemm_bf16_wmma<true, false, true>
        <<<dim3(HDIM / TN, 1, 1), blk, 0, stream>>>(
        g, Wl1, g2, NB, HDIM, HDIM, HDIM, HDIM, HDIM, 0, 0, 0, bl1);
    final_logits<<<(NB * ODIM + 255) / 256, blk, 0, stream>>>(g2, Wl2, bl2, out);
    loss_reduce<<<1, 64, 0, stream>>>(lp, out);
}
